// PhysicsAwareGNN_86182813762316
// MI455X (gfx1250) — matrix operand order repaired and checked by reading, not compiled, not run on hardware
//
#include <hip/hip_runtime.h>
#include <stdint.h>

#define N_NODES 100000
#define E_EDGES 1600000
#define H_DIM   128
#define P_DIM   10
#define TOT     138
#define KPAD    160   // TOT padded up to a multiple of 32

typedef __attribute__((ext_vector_type(16))) _Float16 v16h;
typedef __attribute__((ext_vector_type(8)))  float    v8f;

union Frag {                 // one WMMA A/B operand: 16 halves = 8 VGPRs
    uint4    q[2];
    v16h     v;
    _Float16 h[16];
};
union Acc8 {                 // 16x16 f32 C/D tile: 8 VGPRs
    v8f   v;
    float f[8];
};
union H8 { uint4 q; _Float16 h[8]; };
union H4 { uint2 q; _Float16 h[4]; };

// ---------------------------------------------------------------------------
// Packing kernels: fp32 -> f16, with zero padding
// ---------------------------------------------------------------------------
__global__ __launch_bounds__(256) void pack_fr(const float* __restrict__ phys,
                                               const float* __restrict__ emb,
                                               _Float16* __restrict__ fr)
{
    int idx = blockIdx.x * 256 + threadIdx.x;
    if (idx >= N_NODES * KPAD) return;
    int m = idx / KPAD, c = idx % KPAD;
    float v = 0.f;
    if (c < P_DIM)      v = phys[m * P_DIM + c];
    else if (c < TOT)   v = emb[m * H_DIM + (c - P_DIM)];
    fr[idx] = (_Float16)v;
}

// Wp is [KPAD][128]; row r takes W[r-row0] if (r-row0) in [0, rows_in), else 0
__global__ __launch_bounds__(256) void pack_w(const float* __restrict__ W,
                                              _Float16* __restrict__ Wp,
                                              int rows_in, int row0)
{
    int idx = blockIdx.x * 256 + threadIdx.x;
    if (idx >= KPAD * H_DIM) return;
    int r = idx / H_DIM, c = idx % H_DIM;
    int ri = r - row0;
    float v = (ri >= 0 && ri < rows_in) ? W[ri * H_DIM + c] : 0.f;
    Wp[idx] = (_Float16)v;
}

__global__ __launch_bounds__(256) void pack_w128(const float* __restrict__ W,
                                                 _Float16* __restrict__ Wp)
{
    int idx = blockIdx.x * 256 + threadIdx.x;
    if (idx >= H_DIM * H_DIM) return;
    Wp[idx] = (_Float16)W[idx];
}

// ---------------------------------------------------------------------------
// QKV GEMM: [16 rows] x fr16[N,160] @ {Wq,Wk,Wv}[160,128] via WMMA f16
// block = 256 threads = 8 waves, wave w owns output col-tile [16w, 16w+16)
// ---------------------------------------------------------------------------
__global__ __launch_bounds__(256) void qkv_wmma(
    const _Float16* __restrict__ fr,
    const _Float16* __restrict__ Wq16, const _Float16* __restrict__ Wk16,
    const _Float16* __restrict__ Wv16,
    const float* __restrict__ bq, const float* __restrict__ bk,
    const float* __restrict__ bv,
    _Float16* __restrict__ Qo, _Float16* __restrict__ Ko,
    _Float16* __restrict__ Vo)
{
    const int lane  = threadIdx.x & 31;
    const int wave  = threadIdx.x >> 5;       // n-tile 0..7
    const int row0  = blockIdx.x * 16;
    const int lrow  = lane & 15;              // A: M row / C: N col
    const int khalf = lane >> 4;              // A: K-half selector / C: M-half

    // Preload A fragments for all 5 K-chunks (layout: ISA 7.12.2, 16-bit A 16x32)
    Frag a[5];
    const _Float16* arow = fr + (size_t)(row0 + lrow) * KPAD;
#pragma unroll
    for (int kc = 0; kc < 5; ++kc) {
        int k0 = kc * 32 + khalf * 8;
        a[kc].q[0] = *(const uint4*)(arow + k0);        // K = k0..k0+7
        a[kc].q[1] = *(const uint4*)(arow + k0 + 16);   // K = k0+16..k0+23
    }

    const _Float16* Ws[3] = {Wq16, Wk16, Wv16};
    const float*    bs[3] = {bq, bk, bv};
    _Float16*       Os[3] = {Qo, Ko, Vo};

#pragma unroll
    for (int mat = 0; mat < 3; ++mat) {
        Acc8 acc;
#pragma unroll
        for (int r = 0; r < 8; ++r) acc.f[r] = 0.f;
        const _Float16* W = Ws[mat];
#pragma unroll
        for (int kc = 0; kc < 5; ++kc) {
            Frag b;                             // B 32x16: lane = N col, element s = K row
            const _Float16* bp = W + (size_t)(kc * 32 + khalf * 8) * H_DIM + wave * 16 + lrow;
#pragma unroll
            for (int s = 0; s < 16; ++s)
                b.h[s] = bp[(size_t)(s < 8 ? s : s + 8) * H_DIM];
            acc.v = __builtin_amdgcn_wmma_f32_16x16x32_f16(
                false, a[kc].v, false, b.v, (short)0, acc.v, false, false);
        }
        int   n    = wave * 16 + lrow;
        float bias = bs[mat][n];
        _Float16* O = Os[mat];
#pragma unroll
        for (int r = 0; r < 8; ++r) {           // C layout: M = r + 8*khalf
            int m = r + 8 * khalf;
            O[(size_t)(row0 + m) * H_DIM + n] = (_Float16)(acc.f[r] + bias);
        }
    }
}

// ---------------------------------------------------------------------------
// Edge scores: thread per edge; dot(Q[dst],K[src]) in f32 from f16, exp,
// atomic add into denom[dst]
// ---------------------------------------------------------------------------
__global__ __launch_bounds__(256) void edge_score(
    const _Float16* __restrict__ Q, const _Float16* __restrict__ K,
    const int* __restrict__ ei, float* __restrict__ exps,
    float* __restrict__ denom)
{
    int e = blockIdx.x * 256 + threadIdx.x;
    if (e >= E_EDGES) return;
    int s = ei[e];
    int d = ei[E_EDGES + e];
    const _Float16* qp = Q + (size_t)d * H_DIM;
    const _Float16* kp = K + (size_t)s * H_DIM;
    float dot = 0.f;
#pragma unroll
    for (int i = 0; i < H_DIM; i += 8) {
        H8 qa, ka;
        qa.q = *(const uint4*)(qp + i);
        ka.q = *(const uint4*)(kp + i);
#pragma unroll
        for (int j = 0; j < 8; ++j)
            dot += (float)qa.h[j] * (float)ka.h[j];
    }
    float ex = __expf(dot * 0.08838834764831845f);  // 1/sqrt(128)
    exps[e] = ex;
    unsafeAtomicAdd(&denom[d], ex);
}

// ---------------------------------------------------------------------------
// Message scatter: wave per edge, 4 cols per lane, fp32 atomics into agg
// ---------------------------------------------------------------------------
__global__ __launch_bounds__(256) void edge_msg(
    const _Float16* __restrict__ V, const int* __restrict__ ei,
    const float* __restrict__ exps, const float* __restrict__ denom,
    float* __restrict__ agg)
{
    int lane = threadIdx.x & 31;
    int e    = blockIdx.x * 8 + (threadIdx.x >> 5);
    if (e >= E_EDGES) return;
    int s = ei[e];
    int d = ei[E_EDGES + e];
    float prob = exps[e] / (denom[d] + 1e-8f);
    H4 va;
    va.q = *(const uint2*)(V + (size_t)s * H_DIM + lane * 4);
    float* ap = agg + (size_t)d * H_DIM + lane * 4;
    unsafeAtomicAdd(ap + 0, (float)va.h[0] * prob);
    unsafeAtomicAdd(ap + 1, (float)va.h[1] * prob);
    unsafeAtomicAdd(ap + 2, (float)va.h[2] * prob);
    unsafeAtomicAdd(ap + 3, (float)va.h[3] * prob);
}

// ---------------------------------------------------------------------------
// out = LN(emb + agg @ Wo + bo): WMMA GEMM fused with residual + LayerNorm
// block = 16 rows x 8 waves; LN: wave handles 2 rows via shfl_xor (wave32)
// ---------------------------------------------------------------------------
__global__ __launch_bounds__(256) void out_ln(
    const float* __restrict__ agg, const _Float16* __restrict__ Wo16,
    const float* __restrict__ bo, const float* __restrict__ emb,
    const float* __restrict__ gamma, const float* __restrict__ beta,
    float* __restrict__ out)
{
    __shared__ float xs[16 * 128];
    const int lane  = threadIdx.x & 31;
    const int wave  = threadIdx.x >> 5;
    const int row0  = blockIdx.x * 16;
    const int lrow  = lane & 15;
    const int khalf = lane >> 4;

    Acc8 acc;
#pragma unroll
    for (int r = 0; r < 8; ++r) acc.f[r] = 0.f;

    const float* arow = agg + (size_t)(row0 + lrow) * H_DIM;
#pragma unroll
    for (int kc = 0; kc < 4; ++kc) {
        Frag a;
        int k0 = kc * 32 + khalf * 8;
        float4 f0 = *(const float4*)(arow + k0);
        float4 f1 = *(const float4*)(arow + k0 + 4);
        float4 f2 = *(const float4*)(arow + k0 + 16);
        float4 f3 = *(const float4*)(arow + k0 + 20);
        a.h[0]  = (_Float16)f0.x; a.h[1]  = (_Float16)f0.y;
        a.h[2]  = (_Float16)f0.z; a.h[3]  = (_Float16)f0.w;
        a.h[4]  = (_Float16)f1.x; a.h[5]  = (_Float16)f1.y;
        a.h[6]  = (_Float16)f1.z; a.h[7]  = (_Float16)f1.w;
        a.h[8]  = (_Float16)f2.x; a.h[9]  = (_Float16)f2.y;
        a.h[10] = (_Float16)f2.z; a.h[11] = (_Float16)f2.w;
        a.h[12] = (_Float16)f3.x; a.h[13] = (_Float16)f3.y;
        a.h[14] = (_Float16)f3.z; a.h[15] = (_Float16)f3.w;

        Frag b;
        const _Float16* bp = Wo16 + (size_t)(kc * 32 + khalf * 8) * H_DIM + wave * 16 + lrow;
#pragma unroll
        for (int s = 0; s < 16; ++s)
            b.h[s] = bp[(size_t)(s < 8 ? s : s + 8) * H_DIM];
        acc.v = __builtin_amdgcn_wmma_f32_16x16x32_f16(
            false, a.v, false, b.v, (short)0, acc.v, false, false);
    }

    int   n    = wave * 16 + lrow;
    float bias = bo[n];
#pragma unroll
    for (int r = 0; r < 8; ++r) {
        int m = r + 8 * khalf;
        xs[m * 128 + n] = acc.f[r] + bias + emb[(size_t)(row0 + m) * H_DIM + n];
    }
    __syncthreads();

#pragma unroll
    for (int rr = 0; rr < 2; ++rr) {
        int m = wave * 2 + rr;
        float x0 = xs[m * 128 + lane];
        float x1 = xs[m * 128 + lane + 32];
        float x2 = xs[m * 128 + lane + 64];
        float x3 = xs[m * 128 + lane + 96];
        float s  = x0 + x1 + x2 + x3;
        float ss = x0 * x0 + x1 * x1 + x2 * x2 + x3 * x3;
#pragma unroll
        for (int off = 16; off > 0; off >>= 1) {
            s  += __shfl_xor(s, off, 32);
            ss += __shfl_xor(ss, off, 32);
        }
        float mu  = s * (1.f / 128.f);
        float var = ss * (1.f / 128.f) - mu * mu;
        float inv = rsqrtf(var + 1e-5f);
        float* orow = out + (size_t)(row0 + m) * H_DIM;
        orow[lane]      = gamma[lane]      * (x0 - mu) * inv + beta[lane];
        orow[lane + 32] = gamma[lane + 32] * (x1 - mu) * inv + beta[lane + 32];
        orow[lane + 64] = gamma[lane + 64] * (x2 - mu) * inv + beta[lane + 64];
        orow[lane + 96] = gamma[lane + 96] * (x3 - mu) * inv + beta[lane + 96];
    }
}

// ---------------------------------------------------------------------------
extern "C" void kernel_launch(void* const* d_in, const int* in_sizes, int n_in,
                              void* d_out, int out_size, void* d_ws, size_t ws_size,
                              hipStream_t stream)
{
    const float* phys  = (const float*)d_in[0];
    const float* emb   = (const float*)d_in[1];
    const int*   ei    = (const int*)  d_in[2];
    const float* Wq    = (const float*)d_in[3];
    const float* bq    = (const float*)d_in[4];
    const float* Wk    = (const float*)d_in[5];
    const float* bk    = (const float*)d_in[6];
    const float* Wv    = (const float*)d_in[7];
    const float* bv    = (const float*)d_in[8];
    const float* Wo    = (const float*)d_in[9];
    const float* bo    = (const float*)d_in[10];
    const float* gamma = (const float*)d_in[11];
    const float* beta  = (const float*)d_in[12];
    float* out = (float*)d_out;

    char* ws = (char*)d_ws;
    size_t off = 0;
    auto take = [&](size_t bytes) -> char* {
        char* p = ws + off;
        off += (bytes + 255) & ~(size_t)255;
        return p;
    };
    _Float16* fr16  = (_Float16*)take((size_t)N_NODES * KPAD * 2);
    _Float16* Q16   = (_Float16*)take((size_t)N_NODES * H_DIM * 2);
    _Float16* K16   = (_Float16*)take((size_t)N_NODES * H_DIM * 2);
    _Float16* V16   = (_Float16*)take((size_t)N_NODES * H_DIM * 2);
    _Float16* Wq16  = (_Float16*)take((size_t)KPAD * H_DIM * 2);
    _Float16* Wk16  = (_Float16*)take((size_t)KPAD * H_DIM * 2);
    _Float16* Wv16  = (_Float16*)take((size_t)KPAD * H_DIM * 2);
    _Float16* Wo16  = (_Float16*)take((size_t)H_DIM * H_DIM * 2);
    float*    exps  = (float*)take((size_t)E_EDGES * 4);
    float*    denom = (float*)take((size_t)N_NODES * 4);
    float*    agg   = (float*)take((size_t)N_NODES * H_DIM * 4);

    hipMemsetAsync(denom, 0, (size_t)N_NODES * 4, stream);
    hipMemsetAsync(agg,   0, (size_t)N_NODES * H_DIM * 4, stream);

    pack_fr<<<(N_NODES * KPAD + 255) / 256, 256, 0, stream>>>(phys, emb, fr16);
    pack_w <<<(KPAD * H_DIM + 255) / 256, 256, 0, stream>>>(Wq, Wq16, TOT, 0);
    pack_w <<<(KPAD * H_DIM + 255) / 256, 256, 0, stream>>>(Wk, Wk16, TOT, 0);
    pack_w <<<(KPAD * H_DIM + 255) / 256, 256, 0, stream>>>(Wv, Wv16, H_DIM, P_DIM);
    pack_w128<<<(H_DIM * H_DIM + 255) / 256, 256, 0, stream>>>(Wo, Wo16);

    qkv_wmma<<<N_NODES / 16, 256, 0, stream>>>(fr16, Wq16, Wk16, Wv16,
                                               bq, bk, bv, Q16, K16, V16);
    edge_score<<<E_EDGES / 256, 256, 0, stream>>>(Q16, K16, ei, exps, denom);
    edge_msg  <<<E_EDGES / 8,   256, 0, stream>>>(V16, ei, exps, denom, agg);
    out_ln    <<<N_NODES / 16,  256, 0, stream>>>(agg, Wo16, bo, emb,
                                                  gamma, beta, out);
}
